// EdgeRegressionHead_24180665876934
// MI455X (gfx1250) — compile-verified, matrix operand order checked by reading, not run on hardware
//
#include <hip/hip_runtime.h>
#include <hip/hip_bf16.h>
#include <math.h>

typedef __attribute__((ext_vector_type(16))) _Float16 v16h;
typedef __attribute__((ext_vector_type(8)))  float    v8f;

#define Bb 2
#define Kk 512
#define Dd 128
#define Hh 64

// ---------------- workspace layout (bytes) ----------------
#define HH_OFF   0                              // f16 h            [B*K*D]
#define P_OFF    (HH_OFF   + Bb*Kk*Dd*2)        // f32 P = hW1a+cW1d    [B*K*D]
#define Q_OFF    (P_OFF    + Bb*Kk*Dd*4)        // f32 Q = hW1b-cW1d+b1 [B*K*D]
#define W1CT_OFF (Q_OFF    + Bb*Kk*Dd*4)        // f16 W1c^T        [128][128]
#define W2T_OFF  (W1CT_OFF + Dd*Dd*2)           // f16 W2^T         [64][128]
#define WRAW_OFF (W2T_OFF  + Hh*Dd*2)           // f32 raw sigmoid  [B*K*K]

// ---------------- LDS layout for main kernel ----------------
#define SHS 136   // f16 row stride (halfs): 128 + 8 pad -> bank stagger
#define SFS 132   // f32 row stride (floats)
#define X2S 68    // f32 row stride for x2 scratch (64 + 4)

#define SM_HI   0
#define SM_HJ   (SM_HI + 16*SHS*2)
#define SM_W1   (SM_HJ + 16*SHS*2)
#define SM_W2   (SM_W1 + 128*SHS*2)
#define SM_P    (SM_W2 + 64*SHS*2)
#define SM_Q    (SM_P  + 16*SFS*4)
#define SM_B2   (SM_Q  + 16*SFS*4)
#define SM_W3   (SM_B2 + 64*4)
#define SM_B3   (SM_W3 + 64*4)
#define SM_X1   (SM_B3 + 16)
#define SMEM_BYTES (SM_X1 + 8*16*SHS*2)   // 8 waves * 4352B scratch (~110.5 KB)

__device__ __forceinline__ v8f zero8() {
  v8f z;
#pragma unroll
  for (int t = 0; t < 8; ++t) z[t] = 0.0f;
  return z;
}

// Gather a 16x16x32-f16 A/B fragment from an LDS row-major tile.
// ISA 16-bit layout: lanes 0-15 hold K = {0..7, 16..23}, lanes 16-31 hold
// K = {8..15, 24..31} (relative to koff); row = M (A) or N (B).
__device__ __forceinline__ v16h load_frag(const _Float16* base, int row,
                                          int strideH, int koff, int lane) {
  const _Float16* p = base + row * strideH + koff + ((lane & 16) ? 8 : 0);
  v16h f;
#pragma unroll
  for (int t = 0; t < 8; ++t) { f[t] = p[t]; f[t + 8] = p[t + 16]; }
  return f;
}

// ---------------- kernel 1: per-node precompute ----------------
// P[n] = h.W1[0:128] + c.W1[384:386] ; Q[n] = h.W1[128:256] - c.W1[384:386] + b1
__global__ __launch_bounds__(128)
void edge_prep(const float* __restrict__ h, const float* __restrict__ cen,
               const float* __restrict__ W1, const float* __restrict__ b1,
               _Float16* __restrict__ hh, float* __restrict__ P,
               float* __restrict__ Q) {
  const int bk = blockIdx.x;       // node index in [0, B*K)
  const int n  = threadIdx.x;      // output channel
  __shared__ float sh[Dd];
  float hv = h[bk * Dd + n];
  sh[n] = hv;
  __syncthreads();
  hh[bk * Dd + n] = (_Float16)hv;
  float p = 0.f, q = 0.f;
#pragma unroll 4
  for (int d = 0; d < Dd; ++d) {
    float x = sh[d];
    p = fmaf(x, W1[d * Dd + n], p);
    q = fmaf(x, W1[(Dd + d) * Dd + n], q);
  }
  float c0 = cen[bk * 2 + 0], c1 = cen[bk * 2 + 1];
  float cw = c0 * W1[384 * Dd + n] + c1 * W1[385 * Dd + n];
  P[bk * Dd + n] = p + cw;
  Q[bk * Dd + n] = q - cw + b1[n];
}

// ---------------- kernel 2: weight transpose + f16 convert ----------------
__global__ __launch_bounds__(128)
void edge_wt(const float* __restrict__ W1, const float* __restrict__ W2,
             _Float16* __restrict__ w1ct, _Float16* __restrict__ w2t) {
  const int r = blockIdx.x;   // output row of transposed weight (= N)
  const int k = threadIdx.x;  // K index
  w1ct[r * 128 + k] = (_Float16)W1[(256 + k) * Dd + r];    // W1c^T [N][K]
  if (r < Hh) w2t[r * 128 + k] = (_Float16)W2[k * Hh + r]; // W2^T  [N][K]
}

// ---------------- kernel 3: fused pairwise MLP (WMMA) ----------------
__global__ __launch_bounds__(256)
void edge_main(const _Float16* __restrict__ hh, const float* __restrict__ Pg,
               const float* __restrict__ Qg, const _Float16* __restrict__ w1ct,
               const _Float16* __restrict__ w2t, const float* __restrict__ b2,
               const float* __restrict__ W3, const float* __restrict__ b3,
               float* __restrict__ Wraw) {
  extern __shared__ __align__(16) char smem[];
  _Float16* s_hi = (_Float16*)(smem + SM_HI);
  _Float16* s_hj = (_Float16*)(smem + SM_HJ);
  _Float16* s_w1 = (_Float16*)(smem + SM_W1);
  _Float16* s_w2 = (_Float16*)(smem + SM_W2);
  float*    s_P  = (float*)(smem + SM_P);
  float*    s_Q  = (float*)(smem + SM_Q);
  float*    s_b2 = (float*)(smem + SM_B2);
  float*    s_w3 = (float*)(smem + SM_W3);
  float*    s_b3 = (float*)(smem + SM_B3);
  _Float16* s_x1 = (_Float16*)(smem + SM_X1);

  const int bi = blockIdx.z, it = blockIdx.x, jt = blockIdx.y;
  const int i0 = it * 16, j0 = jt * 16;
  const int tid = threadIdx.x, lane = tid & 31, wv = tid >> 5;

  __builtin_prefetch(w1ct, 0, 0);
  __builtin_prefetch(w2t, 0, 0);

  // ---- cooperative stage into LDS ----
  for (int idx = tid; idx < 16 * Dd; idx += 256) {
    int r = idx >> 7, c = idx & 127;
    s_hi[r * SHS + c] = hh[((bi * Kk) + i0 + r) * Dd + c];
    s_hj[r * SHS + c] = hh[((bi * Kk) + j0 + r) * Dd + c];
    s_P[r * SFS + c]  = Pg[((bi * Kk) + i0 + r) * Dd + c];
    s_Q[r * SFS + c]  = Qg[((bi * Kk) + j0 + r) * Dd + c];
  }
  for (int idx = tid; idx < 128 * 128; idx += 256)
    s_w1[(idx >> 7) * SHS + (idx & 127)] = w1ct[idx];
  for (int idx = tid; idx < 64 * 128; idx += 256)
    s_w2[(idx >> 7) * SHS + (idx & 127)] = w2t[idx];
  if (tid < 64) { s_b2[tid] = b2[tid]; s_w3[tid] = W3[tid]; }
  if (tid == 0) s_b3[0] = b3[0];
  __syncthreads();

  // per-wave scratch: x1 (f16 16x128) aliased later by x2 (f32 16x64)
  _Float16* x1s = s_x1 + wv * 16 * SHS;
  float*    x2s = (float*)x1s;

  const int mbase = (lane >> 4) << 3;  // C/D fragment: M = v + mbase
  const int ncol  = lane & 15;         //               N = nt*16 + ncol

  for (int rep = 0; rep < 2; ++rep) {
    const int mt = wv * 2 + rep;       // i-node (local) for this M-tile
    // ---- layer 1: A = h_i(mt) (.) h_j  ;  [16x128] @ W1c[128x128] ----
    v16h afr[4];
#pragma unroll
    for (int ks = 0; ks < 4; ++ks) {
      v16h aj = load_frag(s_hj, ncol, SHS, ks * 32, lane);
      v16h ai = load_frag(s_hi, mt,   SHS, ks * 32, lane);
      afr[ks] = aj * ai;               // packed f16 muls (v_pk_mul_f16)
    }
    v8f acc1[8];
#pragma unroll
    for (int nt = 0; nt < 8; ++nt) acc1[nt] = zero8();
    // ks outer / nt inner: batch-load 8 B frags, then 8 INDEPENDENT wmmas
#pragma unroll
    for (int ks = 0; ks < 4; ++ks) {
      v16h bfr[8];
#pragma unroll
      for (int nt = 0; nt < 8; ++nt)
        bfr[nt] = load_frag(s_w1, nt * 16 + ncol, SHS, ks * 32, lane);
#pragma unroll
      for (int nt = 0; nt < 8; ++nt)
        acc1[nt] = __builtin_amdgcn_wmma_f32_16x16x32_f16(
            false, afr[ks], false, bfr[nt], (short)0, acc1[nt], false, false);
    }
    // epilogue 1: x1 = relu(acc + P[i] + Q[j]) -> f16 scratch
#pragma unroll
    for (int nt = 0; nt < 8; ++nt) {
      int N = nt * 16 + ncol;
      float pv = s_P[mt * SFS + N];
#pragma unroll
      for (int v = 0; v < 8; ++v) {
        int M = mbase + v;             // j-local
        float x = acc1[nt][v] + pv + s_Q[M * SFS + N];
        x1s[M * SHS + N] = (_Float16)fmaxf(x, 0.0f);
      }
    }
    // ---- layer 2: [16x128] @ W2[128x64] ----
    v16h a2[4];
#pragma unroll
    for (int ks = 0; ks < 4; ++ks)
      a2[ks] = load_frag(x1s, ncol, SHS, ks * 32, lane);
    v8f acc2[4];
#pragma unroll
    for (int nt = 0; nt < 4; ++nt) acc2[nt] = zero8();
#pragma unroll
    for (int ks = 0; ks < 4; ++ks) {
      v16h bfr[4];
#pragma unroll
      for (int nt = 0; nt < 4; ++nt)
        bfr[nt] = load_frag(s_w2, nt * 16 + ncol, SHS, ks * 32, lane);
#pragma unroll
      for (int nt = 0; nt < 4; ++nt)
        acc2[nt] = __builtin_amdgcn_wmma_f32_16x16x32_f16(
            false, a2[ks], false, bfr[nt], (short)0, acc2[nt], false, false);
    }
    // epilogue 2: x2 = relu(acc2 + b2) -> f32 scratch (aliases x1s; a2 in regs)
#pragma unroll
    for (int nt = 0; nt < 4; ++nt) {
      int N = nt * 16 + ncol;
      float bv = s_b2[N];
#pragma unroll
      for (int v = 0; v < 8; ++v)
        x2s[(mbase + v) * X2S + N] = fmaxf(acc2[nt][v] + bv, 0.0f);
    }
    // ---- layer 3: all 32 lanes; lane l does row l%16, half of N range ----
    {
      const int row  = lane & 15;
      const int noff = (lane >> 4) * 32;   // 0 or 32
      float acc = 0.0f;
#pragma unroll 8
      for (int n = 0; n < 32; ++n)
        acc += x2s[row * X2S + noff + n] * s_w3[noff + n];
      acc += __shfl_down(acc, 16);         // combine halves (wave32)
      if (lane < 16) {
        float sig = 1.0f / (1.0f + expf(-(acc + s_b3[0])));
        Wraw[((bi * Kk) + i0 + mt) * Kk + j0 + row] = sig;
      }
    }
  }
}

// ---------------- kernel 4: symmetrize + ghost/diag mask ----------------
__global__ __launch_bounds__(256)
void edge_post(const float* __restrict__ Wraw, const float* __restrict__ mask,
               float* __restrict__ out) {
  int idx = blockIdx.x * 256 + threadIdx.x;
  if (idx >= Bb * Kk * Kk) return;
  int b = idx / (Kk * Kk);
  int rem = idx - b * Kk * Kk;
  int i = rem >> 9, j = rem & (Kk - 1);
  float w = 0.5f * (Wraw[idx] + Wraw[(b * Kk + j) * Kk + i]);
  bool zero = (mask[b * Kk + i] < 0.1f) || (mask[b * Kk + j] < 0.1f) || (i == j);
  out[idx] = zero ? 0.0f : w;
}

extern "C" void kernel_launch(void* const* d_in, const int* in_sizes, int n_in,
                              void* d_out, int out_size, void* d_ws, size_t ws_size,
                              hipStream_t stream) {
  (void)in_sizes; (void)n_in; (void)out_size; (void)ws_size;
  const float* h   = (const float*)d_in[0];
  const float* cen = (const float*)d_in[1];
  const float* nm  = (const float*)d_in[2];
  const float* W1  = (const float*)d_in[3];
  const float* b1  = (const float*)d_in[4];
  const float* W2  = (const float*)d_in[5];
  const float* b2  = (const float*)d_in[6];
  const float* W3  = (const float*)d_in[7];
  const float* b3  = (const float*)d_in[8];

  char* ws = (char*)d_ws;
  _Float16* hh   = (_Float16*)(ws + HH_OFF);
  float*    P    = (float*)(ws + P_OFF);
  float*    Q    = (float*)(ws + Q_OFF);
  _Float16* w1ct = (_Float16*)(ws + W1CT_OFF);
  _Float16* w2t  = (_Float16*)(ws + W2T_OFF);
  float*    Wraw = (float*)(ws + WRAW_OFF);

  edge_prep<<<Bb * Kk, 128, 0, stream>>>(h, cen, W1, b1, hh, P, Q);
  edge_wt<<<128, 128, 0, stream>>>(W1, W2, w1ct, w2t);
  dim3 grid(Kk / 16, Kk / 16, Bb);
  edge_main<<<grid, 256, SMEM_BYTES, stream>>>(hh, P, Q, w1ct, w2t, b2, W3, b3, Wraw);
  edge_post<<<(Bb * Kk * Kk + 255) / 256, 256, 0, stream>>>(Wraw, nm, (float*)d_out);
}
